// TritonShortSeqSelfAttentionModule_30966714204875
// MI455X (gfx1250) — matrix operand order repaired and checked by reading, not compiled, not run on hardware
//
#include <hip/hip_runtime.h>
#include <hip/hip_bf16.h>

// ---- CDNA5 WMMA types ----
typedef __attribute__((ext_vector_type(16))) __bf16 v16bf;
typedef __attribute__((ext_vector_type(8)))  float  v8f;

#define WMMA_BF16(A, B, C) \
  __builtin_amdgcn_wmma_f32_16x16x32_bf16(false, (A), false, (B), (short)0, (C), false, false)

__device__ __forceinline__ v16bf frag16(const __bf16* p0, const __bf16* p1) {
  union { uint4 q[2]; v16bf v; } u;
  u.q[0] = *(const uint4*)p0;   // 16B -> ds_load_b128 / global_load_b128
  u.q[1] = *(const uint4*)p1;
  return u.v;
}

// A-matrix fragment (16x32 bf16) from row-major [M][K] (stride rs elems).
// Lane L: row = row0 + L%16, chunks K = k0 + (L/16)*8 and k0 + 16 + (L/16)*8.
__device__ __forceinline__ v16bf load_a(const __bf16* base, int rs, int row0,
                                        int k0, int l16, int half) {
  const __bf16* p = base + (row0 + l16) * rs + k0 + half * 8;
  return frag16(p, p + 16);
}

// B-matrix fragment (32x16 bf16) from B^T stored row-major [N][K] (stride rs).
// Lane L: col n = n0 + L%16, chunks K = k0 + (L/16)*8 and k0 + 16 + (L/16)*8.
__device__ __forceinline__ v16bf load_b(const __bf16* base, int rs, int n0,
                                        int k0, int l16, int half) {
  const __bf16* p = base + (n0 + l16) * rs + k0 + half * 8;
  return frag16(p, p + 16);
}

// D tile (16x16 f32, 8 VGPRs) -> bf16 row-major [32][rs] at (m0,n0).
// Lane L, reg v: row = m0 + v + 8*(L/16), col = n0 + L%16.
__device__ __forceinline__ void store_d(__bf16* base, int rs, int m0, int n0,
                                        int l16, int half, v8f d) {
#pragma unroll
  for (int v = 0; v < 8; ++v)
    base[(m0 + v + half * 8) * rs + n0 + l16] = (__bf16)d[v];  // hw cvt (pk where possible)
}

// ---- Pre-kernel: weights -> bf16 in B^T-friendly layouts (L2 resident) ----
// ws[0..16383]     : WqT [h][d=32][e=128]  (pre-scaled by 1/sqrt(DH))
// ws[16384..32767] : WkT [h][d=32][e=128]
// ws[32768..49151] : WvT [h][d=32][e=128]
// ws[49152..65535] : WoT [h][o=128][d=32]
__global__ void convert_weights(const float* __restrict__ wq, const float* __restrict__ wk,
                                const float* __restrict__ wv, const float* __restrict__ wo,
                                __bf16* __restrict__ ws) {
  int i = blockIdx.x * blockDim.x + threadIdx.x;
  if (i >= 16384) return;
  int h = i >> 12;
  int rem = i & 4095;
  int d = rem >> 7;
  int e = rem & 127;
  int src = (h * 128 + e) * 32 + d;                       // w_[q|k|v] is [h][e][d]
  ws[i]         = (__bf16)(wq[src] * 0.17677669529663688f);  // fold softmax scale into Wq
  ws[16384 + i] = (__bf16)wk[src];
  ws[32768 + i] = (__bf16)wv[src];
  int o  = (i >> 5) & 127;
  int d2 = i & 31;
  ws[49152 + i] = (__bf16)wo[(h * 32 + d2) * 128 + o];    // w_o is [h][d][o]
}

// ---- Main kernel: 1 block = 1 batch, 4 waves = 4 heads ----
// S=32, E=128, H=4, DH=32, O=128
__global__ __launch_bounds__(128) void attn_kernel(const float* __restrict__ x,
                                                   const __bf16* __restrict__ ws,
                                                   float* __restrict__ out) {
  // LDS: XB 32x136 (8704) + per-head {Q,K,Vt,P,Hd} each 32x40 (1280) => 59904 B
  __shared__ __align__(16) __bf16 smem[29952];

  const int tid  = threadIdx.x;
  const int b    = blockIdx.x;
  const int wvid = tid >> 5;       // wave / head id (0..3)
  const int lane = tid & 31;
  const int l16  = lane & 15;
  const int half = lane >> 4;

  __bf16* XB = smem;                         // [32][136]
  __bf16* QS = smem + 4352 + wvid * 6400;    // [32 s][40 d]
  __bf16* KS = QS + 1280;                    // [32 t][40 d]
  __bf16* VT = QS + 2560;                    // [32 d][40 t]
  __bf16* PS = QS + 3840;                    // [32 s][40 t]
  __bf16* HS = QS + 5120;                    // [32 s][40 d]

  // --- stage x[b] (coalesced float4) as bf16 into LDS ---
  const float4* xb4 = (const float4*)(x + (size_t)b * 4096);
#pragma unroll
  for (int it = 0; it < 8; ++it) {
    int idx4 = tid + it * 128;           // 0..1023 float4s of 32x128
    float4 f = xb4[idx4];
    int row = idx4 >> 5;                 // 32 float4 per row
    int c   = (idx4 & 31) << 2;
    union { __bf16 h[4]; uint2 u; } pk;
    pk.h[0] = (__bf16)f.x;
    pk.h[1] = (__bf16)f.y;
    pk.h[2] = (__bf16)f.z;
    pk.h[3] = (__bf16)f.w;
    *(uint2*)(XB + row * 136 + c) = pk.u;
  }
  __syncthreads();

  const __bf16* wqh = ws + wvid * 4096;            // [32 d][128 e]
  const __bf16* wkh = ws + 16384 + wvid * 4096;
  const __bf16* wvh = ws + 32768 + wvid * 4096;

  // --- Q = X @ (s*Wq) ; K = X @ Wk  (M=s32, N=d32, K=e128); A hoisted over ni ---
#pragma unroll
  for (int mi = 0; mi < 2; ++mi) {
    v16bf a[4];
#pragma unroll
    for (int kk = 0; kk < 4; ++kk) a[kk] = load_a(XB, 136, mi * 16, kk * 32, l16, half);
#pragma unroll
    for (int ni = 0; ni < 2; ++ni) {
      v8f q = {}; v8f k = {};
#pragma unroll
      for (int kk = 0; kk < 4; ++kk) {
        q = WMMA_BF16(a[kk], load_b(wqh, 128, ni * 16, kk * 32, l16, half), q);
        k = WMMA_BF16(a[kk], load_b(wkh, 128, ni * 16, kk * 32, l16, half), k);
      }
      store_d(QS, 40, mi * 16, ni * 16, l16, half, q);
      store_d(KS, 40, mi * 16, ni * 16, l16, half, k);
    }
  }

  // --- V^T = WvT @ X^T  (M=d32, N=t32, K=e128); B^T layout = X row-major ---
#pragma unroll
  for (int mi = 0; mi < 2; ++mi) {
    v16bf a[4];
#pragma unroll
    for (int kk = 0; kk < 4; ++kk) a[kk] = load_a(wvh, 128, mi * 16, kk * 32, l16, half);
#pragma unroll
    for (int ni = 0; ni < 2; ++ni) {
      v8f acc = {};
#pragma unroll
      for (int kk = 0; kk < 4; ++kk)
        acc = WMMA_BF16(a[kk], load_b(XB, 136, ni * 16, kk * 32, l16, half), acc);
      store_d(VT, 40, mi * 16, ni * 16, l16, half, acc);
    }
  }

  // --- scores = Q @ K^T (K=d32, one WMMA per tile); kept in registers ---
  v8f sc[2][2];
#pragma unroll
  for (int mi = 0; mi < 2; ++mi) {
    v16bf aq = load_a(QS, 40, mi * 16, 0, l16, half);
#pragma unroll
    for (int ni = 0; ni < 2; ++ni) {
      v8f z = {};
      sc[mi][ni] = WMMA_BF16(aq, load_b(KS, 40, ni * 16, 0, l16, half), z);
    }
  }

  // --- softmax over 32 cols: each row lives across one 16-lane half ---
#pragma unroll
  for (int mi = 0; mi < 2; ++mi) {
#pragma unroll
    for (int v = 0; v < 8; ++v) {
      float x0 = sc[mi][0][v], x1 = sc[mi][1][v];
      float m = fmaxf(x0, x1);
#pragma unroll
      for (int off = 1; off < 16; off <<= 1) m = fmaxf(m, __shfl_xor(m, off, 32));
      float e0 = __expf(x0 - m), e1 = __expf(x1 - m);
      float s = e0 + e1;
#pragma unroll
      for (int off = 1; off < 16; off <<= 1) s += __shfl_xor(s, off, 32);
      float inv = 1.0f / s;
      sc[mi][0][v] = e0 * inv;
      sc[mi][1][v] = e1 * inv;
    }
  }
#pragma unroll
  for (int mi = 0; mi < 2; ++mi)
#pragma unroll
    for (int ni = 0; ni < 2; ++ni)
      store_d(PS, 40, mi * 16, ni * 16, l16, half, sc[mi][ni]);

  // --- Hd = P @ V  (B^T layout = V^T row-major) ---
#pragma unroll
  for (int mi = 0; mi < 2; ++mi) {
    v16bf ap = load_a(PS, 40, mi * 16, 0, l16, half);
#pragma unroll
    for (int ni = 0; ni < 2; ++ni) {
      v8f z = {};
      v8f hd = WMMA_BF16(ap, load_b(VT, 40, ni * 16, 0, l16, half), z);
      store_d(HS, 40, mi * 16, ni * 16, l16, half, hd);
    }
  }

  __syncthreads();  // all heads' Hd visible

  // --- out = sum_h Hd[h] @ Wo[h]; wave handles o-slab [wvid*32, wvid*32+32) ---
  v8f oacc[2][2];
#pragma unroll
  for (int mi = 0; mi < 2; ++mi)
#pragma unroll
    for (int ni = 0; ni < 2; ++ni) { v8f z = {}; oacc[mi][ni] = z; }

#pragma unroll
  for (int h = 0; h < 4; ++h) {
    const __bf16* HSh = smem + 4352 + h * 6400 + 5120;
    const __bf16* woh = ws + 49152 + (h * 128 + wvid * 32) * 32;  // [o_local][d]
#pragma unroll
    for (int mi = 0; mi < 2; ++mi) {
      v16bf ah = load_a(HSh, 40, mi * 16, 0, l16, half);
#pragma unroll
      for (int ni = 0; ni < 2; ++ni)
        oacc[mi][ni] = WMMA_BF16(ah, load_b(woh, 32, ni * 16, 0, l16, half), oacc[mi][ni]);
    }
  }

  float* outb = out + (size_t)b * 4096;
#pragma unroll
  for (int mi = 0; mi < 2; ++mi)
#pragma unroll
    for (int ni = 0; ni < 2; ++ni)
#pragma unroll
      for (int v = 0; v < 8; ++v)
        outb[(mi * 16 + v + half * 8) * 128 + wvid * 32 + ni * 16 + l16] = oacc[mi][ni][v];
}

extern "C" void kernel_launch(void* const* d_in, const int* in_sizes, int n_in,
                              void* d_out, int out_size, void* d_ws, size_t ws_size,
                              hipStream_t stream) {
  const float* x  = (const float*)d_in[0];
  const float* wq = (const float*)d_in[1];
  const float* wk = (const float*)d_in[2];
  const float* wv = (const float*)d_in[3];
  const float* wo = (const float*)d_in[4];
  __bf16* ws = (__bf16*)d_ws;                  // needs 131072 bytes
  float* out = (float*)d_out;

  int B = in_sizes[0] / (32 * 128);            // 16384

  convert_weights<<<64, 256, 0, stream>>>(wq, wk, wv, wo, ws);
  attn_kernel<<<B, 128, 0, stream>>>(x, ws, out);
}
